// TransferModel_41867341201735
// MI455X (gfx1250) — matrix operand order repaired and checked by reading, not compiled, not run on hardware
//
#include <hip/hip_runtime.h>

#define N_NODES 50000
#define N_EDGES 600000
#define N_GRAPHS 64
#define D 128
#define DG 128
#define DH 1024
#define N_OUT 150
#define N_OUT_PAD 160

typedef __bf16 bf16;
typedef __attribute__((ext_vector_type(16))) __bf16 v16bf;
typedef __attribute__((ext_vector_type(8)))  __bf16 v8bf;
typedef __attribute__((ext_vector_type(8)))  float  v8f;
typedef __attribute__((ext_vector_type(4)))  unsigned int v4u;
typedef __attribute__((ext_vector_type(8)))  int v8i;
typedef __attribute__((ext_vector_type(4)))  int v4i;

#if defined(__gfx1250__) && __has_builtin(__builtin_amdgcn_tensor_load_to_lds)
#define HAVE_TDM 1
#else
#define HAVE_TDM 0
#endif

__device__ __forceinline__ bf16 f2b(float f) { return (bf16)f; }

// A fragment (16x32 bf16) from row-major LDS tile, row stride ld (elems).
// ISA 7.12.2 layout; each lane's 8-K groups are contiguous -> ds_load_b128.
__device__ __forceinline__ v16bf load_a_frag(const bf16* A, int ld, int lane) {
  int m = lane & 15, h = lane >> 4;
  v16bf a;
#pragma unroll
  for (int v = 0; v < 8; ++v) {
    int k0 = ((v >> 2) << 4) + (h << 3) + ((v & 3) << 1);
    a[2 * v]     = A[m * ld + k0];
    a[2 * v + 1] = A[m * ld + k0 + 1];
  }
  return a;
}

// B fragment (32x16) directly from GLOBAL weights stored K-transposed:
// Bt[col][K], col stride kstride. Lane n half h reads K 8h..8h+7 and 16+8h..16+8h+7.
__device__ __forceinline__ v16bf load_b_frag_g(const bf16* Bt, int kstride, int lane) {
  int n = lane & 15, h = lane >> 4;
  const bf16* p = Bt + n * kstride + h * 8;
  union { v16bf v; v8bf q[2]; } u;
  u.q[0] = *(const v8bf*)(p);
  u.q[1] = *(const v8bf*)(p + 16);
  return u.v;
}

__device__ __forceinline__ v8f wmma_bf16(v16bf a, v16bf b, v8f c) {
  return __builtin_amdgcn_wmma_f32_16x16x32_bf16(false, a, false, b, (short)0, c, false, false);
}

__device__ __forceinline__ unsigned pack2(float v0, float v1) {
  union { bf16 h[2]; unsigned u; } p;
  p.h[0] = f2b(v0); p.h[1] = f2b(v1);
  return p.u;
}

#if HAVE_TDM
// TDM 2D tile load global->LDS, data_size=2B, LDS padding via pad_interval/amount.
// OOB rows (beyond rows_avail) zero-fill in LDS (ISA 8.2).
__device__ __forceinline__ void tdm_load_2d(const void* gsrc, unsigned lds_byte_addr,
                                            unsigned rows_avail, unsigned row_stride_elems,
                                            unsigned tile_w, unsigned tile_h,
                                            unsigned pad_interval_code, unsigned pad_amount_code) {
  unsigned long long ga = (unsigned long long)gsrc;
  v4u g0;
  g0[0] = 1u;                                      // count=1, user descriptor
  g0[1] = lds_byte_addr;                           // lds_addr
  g0[2] = (unsigned)(ga & 0xFFFFFFFFull);          // global_addr[31:0]
  g0[3] = (unsigned)(ga >> 32) | (2u << 30);       // global_addr[56:32] | type=2
  v8i g1;
  g1[0] = (int)((1u << 16) | (1u << 20) | (pad_interval_code << 22) | (pad_amount_code << 25));
  g1[1] = (int)((tile_w & 0xFFFFu) << 16);         // tensor_dim0[15:0] (== tile_w)
  g1[2] = (int)(((tile_w >> 16) & 0xFFFFu) | ((rows_avail & 0xFFFFu) << 16));
  g1[3] = (int)(((rows_avail >> 16) & 0xFFFFu) | ((tile_w & 0xFFFFu) << 16)); // tile_dim0
  g1[4] = (int)(tile_h & 0xFFFFu);                 // tile_dim1, tile_dim2=0
  g1[5] = (int)row_stride_elems;                   // tensor_dim0_stride[31:0]
  g1[6] = 0;
  g1[7] = 0;
  v4i z4 = {0, 0, 0, 0};
  v8i z8 = {0, 0, 0, 0, 0, 0, 0, 0};
  __builtin_amdgcn_tensor_load_to_lds(g0, g1, z4, z4, z8, 0);
}
__device__ __forceinline__ unsigned lds_addr_of(const void* p) {
  return (unsigned)(unsigned long long)p;          // LDS aperture: low 32 bits = LDS addr
}
#endif

// ---------------- kernel 0: zero a float buffer ----------------
__global__ void k_zero(float* p, long n) {
  long i = (long)blockIdx.x * blockDim.x + threadIdx.x;
  if (i < n) p[i] = 0.f;
}

// ---------------- kernel 1: weight prep -> bf16, K-TRANSPOSED [col][K] ----------
__global__ void k_prep(const float* W_msg, const float* W_self, const float* conv_w,
                       const float* t_w, const float* out_w,
                       bf16* Wcat_t, bf16* convw_t, bf16* tw_t, bf16* outw_t) {
  int i = blockIdx.x * blockDim.x + threadIdx.x;
  if (i < 128 * 256) {            // Wcat_t[c][r], r: 0-127 W_self, 128-255 W_msg
    int c = i >> 8, r = i & 255;
    Wcat_t[i] = f2b(r < 128 ? W_self[r * 128 + c] : W_msg[(r - 128) * 128 + c]);
    return;
  }
  i -= 128 * 256;
  if (i < 128 * 128) {            // convw_t[c][r]
    int c = i >> 7, r = i & 127;
    convw_t[i] = f2b(conv_w[r * 128 + c]);
    return;
  }
  i -= 128 * 128;
  if (i < 1024 * 256) {           // tw_t[c][r]
    int c = i >> 8, r = i & 255;
    tw_t[i] = f2b(t_w[r * 1024 + c]);
    return;
  }
  i -= 1024 * 256;
  if (i < N_OUT_PAD * 1024) {     // outw_t[c][r], zero-padded cols 150..159
    int c = i >> 10, r = i & 1023;
    outw_t[i] = f2b(c < N_OUT ? out_w[r * N_OUT + c] : 0.f);
  }
}

// ---------------- kernel 2: edge scatter-add: agg[dst] += x[src] ----------------
__global__ void k_scatter(const float* __restrict__ x, const int* __restrict__ eidx,
                          float* __restrict__ agg) {
  int e = blockIdx.x * 2 + (threadIdx.x >> 7);
  int f = threadIdx.x & 127;
  if (e < N_EDGES) {
    int src = eidx[e];
    int dst = eidx[N_EDGES + e];
    atomicAdd(&agg[(long)dst * D + f], x[(long)src * D + f]);
  }
}

// ---------------- kernel 3: fused node kernel ----------------
__global__ __launch_bounds__(256) void k_node(
    const float* __restrict__ x, const float* __restrict__ agg,
    const int* __restrict__ batch, const float* __restrict__ b_res,
    const float* __restrict__ conv_b, const bf16* __restrict__ Wcat_t,
    const bf16* __restrict__ convw_t, bf16* __restrict__ xc,
    float* __restrict__ pool) {
  __shared__ bf16 As[32 * 264];
  __shared__ bf16 Hs[32 * 136];
  int tid = threadIdx.x, lane = tid & 31, wave = tid >> 5;
  long row0 = (long)blockIdx.x * 32;

  for (int i = tid; i < 32 * 128; i += 256) {   // f32->bf16 pairs
    int r = i >> 7, c2 = (i & 127) * 2;
    long row = row0 + r;
    float v0 = 0.f, v1 = 0.f;
    if (row < N_NODES) {
      if (c2 < 128) { v0 = x[row * 128 + c2];         v1 = x[row * 128 + c2 + 1]; }
      else          { v0 = agg[row * 128 + c2 - 128]; v1 = agg[row * 128 + c2 - 127]; }
    }
    *(unsigned*)&As[r * 264 + c2] = pack2(v0, v1);
  }
  __syncthreads();

  int colb = wave * 16;
  v8f acc0 = {}, acc1 = {};
  const bf16* Bw = Wcat_t + colb * 256;
#pragma unroll
  for (int kc = 0; kc < 256; kc += 32) {
    v16bf b  = load_b_frag_g(Bw + kc, 256, lane);
    v16bf a0 = load_a_frag(As + kc, 264, lane);
    v16bf a1 = load_a_frag(As + 16 * 264 + kc, 264, lane);
    acc0 = wmma_bf16(a0, b, acc0);
    acc1 = wmma_bf16(a1, b, acc1);
  }

  int n = lane & 15, half = lane >> 4;
  int col = colb + n;
  float bias = b_res[col];
#pragma unroll
  for (int r = 0; r < 8; ++r) {
    int m0 = half * 8 + r, m1 = m0 + 16;
    float v0 = acc0[r] + bias; v0 = v0 > 0.f ? v0 : 0.f;
    float v1 = acc1[r] + bias; v1 = v1 > 0.f ? v1 : 0.f;
    Hs[m0 * 136 + col] = f2b(v0);
    Hs[m1 * 136 + col] = f2b(v1);
    long row = row0 + m0;
    if (row < N_NODES) xc[row * 256 + col] = f2b(v0);
    row = row0 + m1;
    if (row < N_NODES) xc[row * 256 + col] = f2b(v1);
  }
  __syncthreads();

  v8f acc2 = {}, acc3 = {};
  const bf16* Bc = convw_t + colb * 128;
#pragma unroll
  for (int kc = 0; kc < 128; kc += 32) {
    v16bf b  = load_b_frag_g(Bc + kc, 128, lane);
    v16bf a0 = load_a_frag(Hs + kc, 136, lane);
    v16bf a1 = load_a_frag(Hs + 16 * 136 + kc, 136, lane);
    acc2 = wmma_bf16(a0, b, acc2);
    acc3 = wmma_bf16(a1, b, acc3);
  }

  float cbias = conv_b[col];
#pragma unroll
  for (int r = 0; r < 8; ++r) {
    int m0 = half * 8 + r, m1 = m0 + 16;
    float v0 = acc2[r] + cbias; v0 = v0 > 0.f ? v0 : 0.f;
    float v1 = acc3[r] + cbias; v1 = v1 > 0.f ? v1 : 0.f;
    long row = row0 + m0;
    if (row < N_NODES) {
      int g = batch[row];
      atomicMax((unsigned int*)&pool[g * DG + col], __float_as_uint(v0));
    }
    row = row0 + m1;
    if (row < N_NODES) {
      int g = batch[row];
      atomicMax((unsigned int*)&pool[g * DG + col], __float_as_uint(v1));
    }
  }
}

// ---------------- kernel 4: xc[:, 128:256] = pool[batch] ----------------
__global__ void k_gather_pool(const int* __restrict__ batch,
                              const float* __restrict__ pool,
                              bf16* __restrict__ xc) {
  long node = blockIdx.x;
  int c = threadIdx.x;
  int g = batch[node];
  xc[node * 256 + 128 + c] = f2b(pool[g * DG + c]);
}

// ---------------- kernel 5: t = relu(xc @ t_w + t_b), bf16 out ----------------
// M=64 x N=256 per block; A tile via TDM (padded LDS rows), B frags from global.
__global__ __launch_bounds__(256) void k_gemm_t(
    const bf16* __restrict__ xc, const bf16* __restrict__ tw_t,
    const float* __restrict__ t_b, bf16* __restrict__ t) {
  __shared__ bf16 As[64 * 264];
  int tid = threadIdx.x, lane = tid & 31, wave = tid >> 5;
  long row0 = (long)blockIdx.x * 64;
  int colbase = blockIdx.y * 256;

#if HAVE_TDM
  if (wave == 0) {
    // 256 elems/row (512B), pad 4 DWORDs every 128 DWORDs -> LDS stride 264
    tdm_load_2d(xc + row0 * 256, lds_addr_of(As),
                (unsigned)(N_NODES - row0), 256, 256, 64, /*pad_int 128dw*/6, /*pad 4dw*/3);
  }
  __builtin_amdgcn_s_wait_tensorcnt(0);
#else
  for (int i = tid; i < 64 * 128; i += 256) {
    int r = i >> 7, c2 = (i & 127) * 2;
    long row = row0 + r;
    unsigned pr = 0;
    if (row < N_NODES) pr = *(const unsigned*)&xc[row * 256 + c2];
    *(unsigned*)&As[r * 264 + c2] = pr;
  }
#endif
  __syncthreads();

  int colw = wave * 32;
  const bf16* B0 = tw_t + (long)(colbase + colw) * 256;
  const bf16* B1 = B0 + 16 * 256;
  v8f acc[4][2] = {};
#pragma unroll
  for (int kc = 0; kc < 256; kc += 32) {
    v16bf b0 = load_b_frag_g(B0 + kc, 256, lane);
    v16bf b1 = load_b_frag_g(B1 + kc, 256, lane);
#pragma unroll
    for (int rt = 0; rt < 4; ++rt) {
      v16bf a = load_a_frag(As + rt * 16 * 264 + kc, 264, lane);
      acc[rt][0] = wmma_bf16(a, b0, acc[rt][0]);
      acc[rt][1] = wmma_bf16(a, b1, acc[rt][1]);
    }
  }

  int n = lane & 15, half = lane >> 4;
#pragma unroll
  for (int ct = 0; ct < 2; ++ct) {
    int col = colbase + colw + ct * 16 + n;
    float bias = t_b[col];
#pragma unroll
    for (int rt = 0; rt < 4; ++rt) {
#pragma unroll
      for (int r = 0; r < 8; ++r) {
        long row = row0 + rt * 16 + half * 8 + r;
        float v = acc[rt][ct][r] + bias; v = v > 0.f ? v : 0.f;
        if (row < N_NODES) t[row * DH + col] = f2b(v);
      }
    }
  }
}

// ---------------- kernel 6: out = t @ out_w + out_b (f32) ----------------
__global__ __launch_bounds__(320) void k_gemm_out(
    const bf16* __restrict__ t, const bf16* __restrict__ outw_t,
    const float* __restrict__ out_b, float* __restrict__ out) {
  __shared__ bf16 As[32 * 40];
  int tid = threadIdx.x, lane = tid & 31, wave = tid >> 5;
  long row0 = (long)blockIdx.x * 32;
  int colb = wave * 16;
  const bf16* Bo = outw_t + colb * 1024;
  v8f acc0 = {}, acc1 = {};

  for (int kc = 0; kc < DH; kc += 32) {
    __syncthreads();   // previous chunk fully consumed before overwrite
#if HAVE_TDM
    if (wave == 0) {
      // 32 elems/row (64B), pad 4 DWORDs every 16 DWORDs -> LDS stride 40
      tdm_load_2d(t + row0 * DH + kc, lds_addr_of(As),
                  (unsigned)(N_NODES - row0), DH, 32, 32, /*pad_int 16dw*/3, /*pad 4dw*/3);
    }
    __builtin_amdgcn_s_wait_tensorcnt(0);
#else
    for (int i = tid; i < 32 * 16; i += 320) {
      int r = i >> 4, c2 = (i & 15) * 2;
      long row = row0 + r;
      unsigned pr = 0;
      if (row < N_NODES) pr = *(const unsigned*)&t[row * DH + kc + c2];
      *(unsigned*)&As[r * 40 + c2] = pr;
    }
#endif
    __syncthreads();
    v16bf b  = load_b_frag_g(Bo + kc, 1024, lane);
    v16bf a0 = load_a_frag(As, 40, lane);
    v16bf a1 = load_a_frag(As + 16 * 40, 40, lane);
    acc0 = wmma_bf16(a0, b, acc0);
    acc1 = wmma_bf16(a1, b, acc1);
  }

  int n = lane & 15, half = lane >> 4;
  int col = colb + n;
  if (col < N_OUT) {
    float bias = out_b[col];
#pragma unroll
    for (int r = 0; r < 8; ++r) {
      int m0 = half * 8 + r, m1 = m0 + 16;
      long row = row0 + m0;
      if (row < N_NODES) out[row * N_OUT + col] = acc0[r] + bias;
      row = row0 + m1;
      if (row < N_NODES) out[row * N_OUT + col] = acc1[r] + bias;
    }
  }
}

extern "C" void kernel_launch(void* const* d_in, const int* in_sizes, int n_in,
                              void* d_out, int out_size, void* d_ws, size_t ws_size,
                              hipStream_t stream) {
  const float* x      = (const float*)d_in[0];
  const int*   eidx   = (const int*)d_in[1];
  const int*   batch  = (const int*)d_in[2];
  const float* W_msg  = (const float*)d_in[3];
  const float* W_self = (const float*)d_in[4];
  const float* b_res  = (const float*)d_in[5];
  const float* conv_w = (const float*)d_in[6];
  const float* conv_b = (const float*)d_in[7];
  const float* t_w    = (const float*)d_in[8];
  const float* t_b    = (const float*)d_in[9];
  const float* out_w  = (const float*)d_in[10];
  const float* out_b  = (const float*)d_in[11];
  float* out = (float*)d_out;

  char* ws = (char*)d_ws;
  size_t off = 0;
  auto alloc = [&](size_t bytes) -> void* {
    void* p = ws + off;
    off += (bytes + 255) & ~(size_t)255;
    return p;
  };
  float* agg    = (float*)alloc((size_t)N_NODES * D * 4);
  float* pool   = (float*)alloc((size_t)N_GRAPHS * DG * 4);
  bf16*  xc     = (bf16*)alloc((size_t)N_NODES * 256 * 2);
  bf16*  tbuf   = (bf16*)alloc((size_t)N_NODES * DH * 2);
  bf16*  Wcat_t = (bf16*)alloc((size_t)128 * 256 * 2);
  bf16*  convw_t= (bf16*)alloc((size_t)128 * 128 * 2);
  bf16*  tw_t   = (bf16*)alloc((size_t)1024 * 256 * 2);
  bf16*  outw_t = (bf16*)alloc((size_t)N_OUT_PAD * 1024 * 2);

  long nz = (long)N_NODES * D;
  k_zero<<<(int)((nz + 255) / 256), 256, 0, stream>>>(agg, nz);
  k_zero<<<(N_GRAPHS * DG + 255) / 256, 256, 0, stream>>>(pool, N_GRAPHS * DG);

  int prep_elems = 128 * 256 + 128 * 128 + 1024 * 256 + N_OUT_PAD * 1024;
  k_prep<<<(prep_elems + 255) / 256, 256, 0, stream>>>(W_msg, W_self, conv_w, t_w, out_w,
                                                       Wcat_t, convw_t, tw_t, outw_t);

  k_scatter<<<N_EDGES / 2, 256, 0, stream>>>(x, eidx, agg);

  int mblocks32 = (N_NODES + 31) / 32;
  k_node<<<mblocks32, 256, 0, stream>>>(x, agg, batch, b_res, conv_b, Wcat_t, convw_t, xc, pool);

  k_gather_pool<<<N_NODES, 128, 0, stream>>>(batch, pool, xc);

  int mblocks64 = (N_NODES + 63) / 64;
  k_gemm_t<<<dim3(mblocks64, DH / 256), 256, 0, stream>>>(xc, tw_t, t_b, tbuf);

  k_gemm_out<<<mblocks32, 320, 0, stream>>>(tbuf, outw_t, out_b, out);
}
